// SparseMoE_cv_9517647528392
// MI455X (gfx1250) — matrix operand order repaired and checked by reading, not compiled, not run on hardware
//
#include <hip/hip_runtime.h>
#include <hip/hip_bf16.h>

// ---------------------------------------------------------------------------
// SparseMoE (conv feature map) for gfx1250 (MI455X).
//   - router fp32 (tiny), top-2; "overwrite" semantics => ONE expert per token
//   - experts: grouped GEMM, bf16 WMMA with f32 accumulate
//   - weights pre-converted to bf16 AND transposed [e][n][k]: a lane's B
//     fragment is one contiguous 32B load (2x global_load_b128)
//   - M-tile = 64 tokens + hidden chunking (8 x 128): cuts per-expert weight
//     re-reads 4x vs a 16-token tile, balancing L2 bytes vs WMMA throughput
// ---------------------------------------------------------------------------

typedef __attribute__((ext_vector_type(16))) __bf16 v16bf;
typedef __attribute__((ext_vector_type(8)))  __bf16 v8bf;
typedef __attribute__((ext_vector_type(8)))  float  v8f;

#define BSZ   16
#define DIMV  256
#define HWV   4096
#define NTOK  (BSZ * HWV)   // 65536 tokens
#define NEXP  8
#define DHID  1024
#define TPB   256

#define MTILE 64            // tokens per FFN block
#define CHID  128           // hidden chunk
#define XPAD  (DIMV + 8)    // 528B rows -> 16B aligned
#define HCP   (CHID + 8)    // 272B rows -> 16B aligned

__device__ __forceinline__ float softplusf(float v) {
    return (v > 20.f) ? v : log1pf(expf(v));
}

__device__ __forceinline__ v16bf cat8(v8bf lo, v8bf hi) {
    return __builtin_shufflevector(lo, hi, 0, 1, 2, 3, 4, 5, 6, 7,
                                           8, 9, 10, 11, 12, 13, 14, 15);
}

// ---- prep A: x (f32) -> xbf (bf16) -----------------------------------------
__global__ void moe_cvt_x_kernel(const float* __restrict__ x, __bf16* __restrict__ xbf) {
    const long i = ((long)blockIdx.x * TPB + threadIdx.x) * 8;
    const float4 a = ((const float4*)(x + i))[0];
    const float4 b = ((const float4*)(x + i))[1];
    v8bf o;
    o[0] = (__bf16)a.x; o[1] = (__bf16)a.y; o[2] = (__bf16)a.z; o[3] = (__bf16)a.w;
    o[4] = (__bf16)b.x; o[5] = (__bf16)b.y; o[6] = (__bf16)b.z; o[7] = (__bf16)b.w;
    *(v8bf*)(xbf + i) = o;
}

// ---- prep B: W[e][k][n] (f32) -> Wt[e][n][k] (bf16), tiled transpose -------
__global__ void moe_w_transpose_kernel(const float* __restrict__ src,
                                       __bf16* __restrict__ dst, int K, int N) {
    __shared__ float tile[32][33];
    const int e = blockIdx.z;
    src += (long)e * K * N;
    dst += (long)e * K * N;
    const int k0 = blockIdx.y * 32, n0 = blockIdx.x * 32;
    for (int r = threadIdx.y; r < 32; r += 8)
        tile[r][threadIdx.x] = src[(long)(k0 + r) * N + n0 + threadIdx.x];
    __syncthreads();
    for (int r = threadIdx.y; r < 32; r += 8)
        dst[(long)(n0 + r) * K + k0 + threadIdx.x] = (__bf16)tile[threadIdx.x][r];
}

// ---- x_avg[b,d] = mean over contiguous 4096 elems --------------------------
__global__ void moe_avg_kernel(const float* __restrict__ x, float* __restrict__ xavg) {
    __shared__ float red[TPB];
    const int row = blockIdx.x;
    const float* p = x + (long)row * HWV;
    float s = 0.f;
    for (int i = threadIdx.x; i < HWV; i += TPB) s += p[i];
    red[threadIdx.x] = s;
    __syncthreads();
    for (int off = TPB / 2; off > 0; off >>= 1) {
        if (threadIdx.x < off) red[threadIdx.x] += red[threadIdx.x + off];
        __syncthreads();
    }
    if (threadIdx.x == 0) xavg[row] = red[0] * (1.f / (float)HWV);
}

// ---- alog[b,e] = x_avg[b]·Wa[:,e] + ba[e]; zero expert counters ------------
__global__ void moe_alogit_kernel(const float* __restrict__ xavg,
                                  const float* __restrict__ Wa,
                                  const float* __restrict__ ba,
                                  float* __restrict__ alog,
                                  int* __restrict__ counts) {
    const int t = threadIdx.x;
    if (t < NEXP) counts[t] = 0;
    if (t >= BSZ * NEXP) return;
    const int b = t >> 3, e = t & 7;
    float s = ba[e];
    for (int d = 0; d < DIMV; ++d)
        s = fmaf(xavg[b * DIMV + d], Wa[d * NEXP + e], s);
    alog[t] = s;
}

// ---- router: noisy top-2, gate of max-index expert, compaction -------------
__global__ void moe_router_kernel(const float* __restrict__ x,
                                  const float* __restrict__ noise,
                                  const float* __restrict__ Wr,
                                  const float* __restrict__ br,
                                  const float* __restrict__ Wn,
                                  const float* __restrict__ bn,
                                  const float* __restrict__ alog,
                                  float* __restrict__ gates,
                                  int* __restrict__ counts,
                                  int* __restrict__ lists) {
    __shared__ float sWr[DIMV * NEXP];
    __shared__ float sWn[DIMV * NEXP];
    const int t = threadIdx.x;
    for (int i = t; i < DIMV * NEXP; i += TPB) { sWr[i] = Wr[i]; sWn[i] = Wn[i]; }
    __syncthreads();

    const int tok = blockIdx.x * TPB + t;
    const int b = tok >> 12;
    const float* xr = x + (long)tok * DIMV;

    float aR[NEXP] = {}, aN[NEXP] = {};
    const float4* x4 = (const float4*)xr;
    for (int d4 = 0; d4 < DIMV / 4; ++d4) {
        float4 xv = x4[d4];
        const float xs[4] = {xv.x, xv.y, xv.z, xv.w};
#pragma unroll
        for (int q = 0; q < 4; ++q) {
            const int d = d4 * 4 + q;
#pragma unroll
            for (int e = 0; e < NEXP; ++e) {
                aR[e] = fmaf(xs[q], sWr[d * NEXP + e], aR[e]);
                aN[e] = fmaf(xs[q], sWn[d * NEXP + e], aN[e]);
            }
        }
    }

    float noisy[NEXP];
#pragma unroll
    for (int e = 0; e < NEXP; ++e) {
        const float sp = softplusf(aN[e] + bn[e]);
        noisy[e] = aR[e] + br[e] + alog[b * NEXP + e]
                 + noise[(long)tok * NEXP + e] * sp;
    }

    float v0 = -3.4e38f; int i0 = 0;
#pragma unroll
    for (int e = 0; e < NEXP; ++e) if (noisy[e] > v0) { v0 = noisy[e]; i0 = e; }
    float v1 = -3.4e38f; int i1 = 0;
#pragma unroll
    for (int e = 0; e < NEXP; ++e) if (e != i0 && noisy[e] > v1) { v1 = noisy[e]; i1 = e; }

    const int   ec = (i0 > i1) ? i0 : i1;           // later expert overwrites
    const float vc = (ec == i0) ? v0 : v1;
    const float vo = (ec == i0) ? v1 : v0;
    const float gate = 1.f / (1.f + expf(vo - vc)); // softmax over top-2

    gates[tok] = gate;
    const int pos = atomicAdd(&counts[ec], 1);
    lists[ec * NTOK + pos] = tok;
}

// ---- grouped expert FFN: 64-token tile, hidden-chunked, bf16 WMMA ----------
__global__ void __launch_bounds__(TPB)
moe_ffn_kernel(const __bf16* __restrict__ xbf,
               const __bf16* __restrict__ W1t, const float* __restrict__ b1,
               const __bf16* __restrict__ W2t, const float* __restrict__ b2,
               const int*   __restrict__ counts, const int* __restrict__ lists,
               const float* __restrict__ gates, float* __restrict__ out) {
    const int e    = blockIdx.y;
    const int cnt  = counts[e];
    const int base = blockIdx.x * MTILE;
    if (base >= cnt) return;
    const int mrows = min(MTILE, cnt - base);

    __shared__ __bf16 xa[MTILE * XPAD];   // 33 KB: 64 token rows (bf16)
    __shared__ __bf16 hc[MTILE * HCP];    // 17 KB: 64 x 128 hidden chunk (bf16)
    __shared__ int    toks[MTILE];
    __shared__ float  grow[MTILE];

    const int t = threadIdx.x;
    if (t < MTILE) {
        const int tok = (t < mrows) ? lists[e * NTOK + base + t] : -1;
        toks[t] = tok;
        grow[t] = (tok >= 0) ? gates[tok] : 0.f;
    }
    __syncthreads();

    // stage 64 bf16 token rows with 16B vector copies
    const uint4* xb4 = (const uint4*)xbf;             // 32 uint4 per row
    for (int idx = t; idx < MTILE * (DIMV / 8); idx += TPB) {
        const int r = idx >> 5, c = idx & 31;
        const int tok = toks[r];
        uint4 v = {0u, 0u, 0u, 0u};
        if (tok >= 0) v = xb4[(long)tok * (DIMV / 8) + c];
        ((uint4*)&xa[r * XPAD])[c] = v;
    }

    const int wave = t >> 5;
    const int lane = t & 31;
    const int l16  = lane & 15;
    const int hi   = lane >> 4;

    const __bf16* W1te = W1t + (long)e * DIMV * DHID;
    const __bf16* W2te = W2t + (long)e * DHID * DIMV;

    v8f oacc[4][2] = {};   // persistent out accumulators: 4 M-tiles x 2 N-tiles

    for (int c = 0; c < DHID / CHID; ++c) {
        // ---- GEMM1 chunk: wave owns N-tile n1 (16 cols) x all 4 M-tiles ----
        // one B fragment load per K-step feeds 4 WMMAs
        const int n1 = c * CHID + wave * 16;
        v8f acc1[4] = {};
        for (int ks = 0; ks < DIMV / 32; ++ks) {
            const int kb = ks * 32;
            const __bf16* bptr = W1te + (long)(n1 + l16) * DIMV + kb + hi * 16;
            __builtin_prefetch(bptr + 32, 0, 1);      // next K slab
            const v16bf bm = *(const v16bf*)bptr;
#pragma unroll
            for (int mt = 0; mt < 4; ++mt) {
                const int row = mt * 16 + l16;
                const v8bf alo = *(const v8bf*)&xa[row * XPAD + kb + hi * 16];
                const v8bf ahi = *(const v8bf*)&xa[row * XPAD + kb + hi * 16 + 8];
                acc1[mt] = __builtin_amdgcn_wmma_f32_16x16x32_bf16(
                               false, cat8(alo, ahi), false, bm,
                               (short)0, acc1[mt], false, false);
            }
        }
        __syncthreads();   // staging done (c==0) / prev GEMM2 done reading hc
        const float bias1 = b1[(long)e * DHID + n1 + l16];
#pragma unroll
        for (int mt = 0; mt < 4; ++mt)
#pragma unroll
            for (int r = 0; r < 8; ++r) {
                float h = acc1[mt][r] + bias1;
                h = h > 0.f ? h : 0.f;
                hc[(mt * 16 + r + 8 * hi) * HCP + wave * 16 + l16] = (__bf16)h;
            }
        __syncthreads();

        // ---- GEMM2 partial: oacc += hc @ W2t rows k in [c*128, +128) -------
        // wave owns out cols n = wave*32 .. +31 (2 N-tiles) x all 4 M-tiles
        for (int ks = 0; ks < CHID / 32; ++ks) {
            const int kb = ks * 32;
            const __bf16* b2base = W2te + (long)(wave * 32 + l16) * DHID
                                        + c * CHID + kb + hi * 16;
            __builtin_prefetch(b2base + 32, 0, 1);
            const v16bf bm0 = *(const v16bf*)b2base;
            const v16bf bm1 = *(const v16bf*)(b2base + (long)16 * DHID);
#pragma unroll
            for (int mt = 0; mt < 4; ++mt) {
                const int row = mt * 16 + l16;
                const v8bf alo = *(const v8bf*)&hc[row * HCP + kb + hi * 16];
                const v8bf ahi = *(const v8bf*)&hc[row * HCP + kb + hi * 16 + 8];
                const v16bf a = cat8(alo, ahi);
                oacc[mt][0] = __builtin_amdgcn_wmma_f32_16x16x32_bf16(
                                  false, a, false, bm0, (short)0, oacc[mt][0], false, false);
                oacc[mt][1] = __builtin_amdgcn_wmma_f32_16x16x32_bf16(
                                  false, a, false, bm1, (short)0, oacc[mt][1], false, false);
            }
        }
    }

    // ---- epilogue: gate-scale + bias, scatter 64 token rows ----------------
#pragma unroll
    for (int j = 0; j < 2; ++j) {
        const int nb = wave * 32 + j * 16;
        const float bias2 = b2[(long)e * DIMV + nb + l16];
#pragma unroll
        for (int mt = 0; mt < 4; ++mt)
#pragma unroll
            for (int r = 0; r < 8; ++r) {
                const int m = mt * 16 + r + 8 * hi;
                if (m < mrows) {
                    out[(long)toks[m] * DIMV + nb + l16] =
                        oacc[mt][j][r] * grow[m] + bias2;
                }
            }
    }
}

// ---------------------------------------------------------------------------
extern "C" void kernel_launch(void* const* d_in, const int* in_sizes, int n_in,
                              void* d_out, int out_size, void* d_ws, size_t ws_size,
                              hipStream_t stream) {
    (void)in_sizes; (void)n_in; (void)out_size; (void)ws_size;

    const float* x     = (const float*)d_in[0];
    const float* noise = (const float*)d_in[1];
    const float* Wr    = (const float*)d_in[2];
    const float* br    = (const float*)d_in[3];
    const float* Wa    = (const float*)d_in[4];
    const float* ba    = (const float*)d_in[5];
    const float* Wn    = (const float*)d_in[6];
    const float* bn    = (const float*)d_in[7];
    const float* W1    = (const float*)d_in[8];
    const float* b1    = (const float*)d_in[9];
    const float* W2    = (const float*)d_in[10];
    const float* b2    = (const float*)d_in[11];
    float* out = (float*)d_out;

    // ---- workspace layout ----
    float* f      = (float*)d_ws;
    float* xavg   = f;                        // 4096 f32
    float* alog   = f + 4096;                 // 128 f32 (padded to 4096)
    float* gates  = f + 8192;                 // NTOK f32
    int*   counts = (int*)(f + 8192 + NTOK);  // 16 ints
    int*   lists  = counts + 16;              // NEXP*NTOK ints
    __bf16* xbf = (__bf16*)(lists + NEXP * NTOK);          // NTOK*DIMV bf16
    __bf16* W1t = xbf + (long)NTOK * DIMV;                 // NEXP*DIMV*DHID bf16
    __bf16* W2t = W1t + (long)NEXP * DIMV * DHID;          // NEXP*DHID*DIMV bf16

    // ---- prep: bf16 conversions / transposes ----
    moe_cvt_x_kernel<<<(NTOK * DIMV) / (TPB * 8), TPB, 0, stream>>>(x, xbf);
    moe_w_transpose_kernel<<<dim3(DHID / 32, DIMV / 32, NEXP), dim3(32, 8),
                             0, stream>>>(W1, W1t, DIMV, DHID);
    moe_w_transpose_kernel<<<dim3(DIMV / 32, DHID / 32, NEXP), dim3(32, 8),
                             0, stream>>>(W2, W2t, DHID, DIMV);

    // ---- router ----
    moe_avg_kernel<<<BSZ * DIMV, TPB, 0, stream>>>(x, xavg);
    moe_alogit_kernel<<<1, 128, 0, stream>>>(xavg, Wa, ba, alog, counts);
    moe_router_kernel<<<NTOK / TPB, TPB, 0, stream>>>(x, noise, Wr, br, Wn, bn,
                                                      alog, gates, counts, lists);

    // ---- grouped expert FFN (bf16 WMMA) ----
    moe_ffn_kernel<<<dim3(NTOK / MTILE, NEXP), TPB, 0, stream>>>(xbf, W1t, b1, W2t, b2,
                                                                 counts, lists, gates, out);
}
